// SlidingWindowAttention_65687229825768
// MI455X (gfx1250) — matrix operand order repaired and checked by reading, not compiled, not run on hardware
//
#include <hip/hip_runtime.h>
#include <hip/hip_bf16.h>

#define HN 16
#define TT 4096
#define DD 1024
#define HD 64
#define BB 2
#define WINSZ 512
#define BSZ 64
#define NBLK (TT / BSZ)          // 64 query blocks per (b,h)
#define NW1 (WINSZ / BSZ + 1)    // 9 key blocks touching a query block
#define SKEY (NW1 * BSZ)         // 576 gathered keys per query block

#define GEMM_BM 256              // rows per workgroup (8 waves x 32)
#define GEMM_BN 64               // cols per workgroup
#define GEMM_BK 32               // k-step (one bf16 WMMA K)
#define GEMM_KSTEPS (DD / GEMM_BK)
#define BROW 40                  // padded LDS row stride (elements) for B tile

typedef __attribute__((ext_vector_type(16))) __bf16 v16bf;
typedef __attribute__((ext_vector_type(8)))  __bf16 v8bf;
typedef __attribute__((ext_vector_type(8)))  float  v8f;
typedef __attribute__((ext_vector_type(4)))  int    v4i;

union AB16 { v16bf v; v8bf h[2]; };

#if defined(__gfx1250__) &&                                                   \
    __has_builtin(__builtin_amdgcn_global_load_async_to_lds_b128) &&          \
    __has_builtin(__builtin_amdgcn_s_wait_asynccnt)
#define USE_ASYNC_LDS 1
typedef __attribute__((address_space(1))) void gvoid;
typedef __attribute__((address_space(3))) void lvoid;
typedef __attribute__((address_space(1))) v4i  gv4i;
typedef __attribute__((address_space(3))) v4i  lv4i;
#else
#define USE_ASYNC_LDS 0
#endif

__device__ __forceinline__ v8f wmma_bf16(v16bf a, v16bf b, v8f c) {
  // 8 args: (neg_a, A, neg_b, B, c_mod, C, reuse_a, reuse_b)
  return __builtin_amdgcn_wmma_f32_16x16x32_bf16(false, a, false, b, (short)0, c,
                                                 false, false);
}

__device__ __forceinline__ v8f vzero8() {
  v8f z = {0.f, 0.f, 0.f, 0.f, 0.f, 0.f, 0.f, 0.f};
  return z;
}

// 16-byte global -> LDS stage: async DMA (ASYNCcnt) if available, else via regs
__device__ __forceinline__ void stage16(const __bf16* src, __bf16* dst) {
#if USE_ASYNC_LDS
  __builtin_amdgcn_global_load_async_to_lds_b128((gv4i*)(gvoid*)src,
                                                 (lv4i*)(lvoid*)dst, 0, 0);
#else
  *(v8bf*)dst = *(const v8bf*)src;
#endif
}

template <int N>
__device__ __forceinline__ void wait_async() {
#if USE_ASYNC_LDS
  __builtin_amdgcn_s_wait_asynccnt(N);
#endif
}

// ---------------------------------------------------------------------------
// Kernel 0: f32 -> bf16 conversion (grid-stride)
// ---------------------------------------------------------------------------
__global__ __launch_bounds__(256, 1) void cvt_f32_bf16(
    const float* __restrict__ in, __bf16* __restrict__ out, int n) {
  int i = blockIdx.x * blockDim.x + threadIdx.x;
  int stride = gridDim.x * blockDim.x;
  for (; i < n; i += stride) out[i] = (__bf16)in[i];
}

// ---------------------------------------------------------------------------
// Shared GEMM mainloop: C(256x64) = X-rows @ W^T, B tile staged in LDS via
// async global->LDS DMA (double buffered), A streamed from global per wave.
//   acc[mt][nt]: wave's 32x64 result (2 M-tiles x 4 N-tiles).
// ---------------------------------------------------------------------------
__device__ __forceinline__ void gemm_mainloop_256x64(
    const __bf16* __restrict__ Xbase,   // (rows x DD) row-major bf16
    const __bf16* __restrict__ W,       // (DD x DD) row-major bf16
    int mbase, int nbase,
    __bf16* bsm /* [2][GEMM_BN*BROW] */,
    v8f acc[2][4]) {
  const int tid  = threadIdx.x;
  const int lane = tid & 31;
  const int hi   = lane >> 4;
  const int lo   = lane & 15;

  // Cooperative B-tile stage: 256 threads x 16B = 4KB = 64 rows x 64B.
  const int bj  = tid >> 2;       // B row (output feature) 0..63
  const int bch = tid & 3;        // 16-byte chunk 0..3
  const __bf16* bsrc_row = W + (size_t)(nbase + bj) * DD + bch * 8;
  __bf16* bdst = bsm + bj * BROW + bch * 8;

#pragma unroll
  for (int mt = 0; mt < 2; ++mt)
#pragma unroll
    for (int nt = 0; nt < 4; ++nt) acc[mt][nt] = vzero8();

  const __bf16* arow0 = Xbase + (size_t)(mbase + lo) * DD;
  const __bf16* arow1 = Xbase + (size_t)(mbase + 16 + lo) * DD;

  stage16(bsrc_row, bdst);  // prologue: buffer 0

#pragma unroll 1            // keep one k-step live: no spill of C tiles
  for (int ks = 0; ks < GEMM_KSTEPS; ++ks) {
    const int kb = ks * GEMM_BK;
    if (ks + 1 < GEMM_KSTEPS) {
      stage16(bsrc_row + kb + GEMM_BK,
              bdst + (((ks + 1) & 1) * (GEMM_BN * BROW)));
      wait_async<1>();  // current buffer's DMA complete (next one in flight)
    } else {
      wait_async<0>();  // last buffer's DMA complete
    }
    __syncthreads();    // all waves' DMAs for this buffer are done

    AB16 a0, a1;  // A-layout: 16 contiguous K at offset hi*16
    a0.h[0] = *(const v8bf*)(arow0 + kb + hi * 16);
    a0.h[1] = *(const v8bf*)(arow0 + kb + hi * 16 + 8);
    a1.h[0] = *(const v8bf*)(arow1 + kb + hi * 16);
    a1.h[1] = *(const v8bf*)(arow1 + kb + hi * 16 + 8);

    const __bf16* bb = bsm + (ks & 1) * (GEMM_BN * BROW);
#pragma unroll
    for (int nt = 0; nt < 4; ++nt) {
      // B-layout: lane col = lo, 16 contiguous K at offset hi*16
      v16bf bv = *(const v16bf*)(bb + (nt * 16 + lo) * BROW + hi * 16);
      acc[0][nt] = wmma_bf16(a0.v, bv, acc[0][nt]);
      acc[1][nt] = wmma_bf16(a1.v, bv, acc[1][nt]);
    }
    __syncthreads();    // readers done before buffer is overwritten
  }
}

// ---------------------------------------------------------------------------
// Kernel 1: fused QKV projection.  grid (32, 16, 3), block 256.
// q,k stored (B,H,T,hd) bf16; v stored transposed (B,H,hd,T) bf16.
// ---------------------------------------------------------------------------
__global__ __launch_bounds__(256, 1) void qkv_gemm(
    const __bf16* __restrict__ xbf,
    const __bf16* __restrict__ wq,
    const __bf16* __restrict__ wk,
    const __bf16* __restrict__ wv,
    __bf16* __restrict__ qws,
    __bf16* __restrict__ kws,
    __bf16* __restrict__ vws) {
  __shared__ __bf16 bsm[2 * GEMM_BN * BROW];
  const int lane = threadIdx.x & 31;
  const int wave = threadIdx.x >> 5;
  const int hi   = lane >> 4;
  const int lo   = lane & 15;
  const int mbase = blockIdx.x * GEMM_BM + wave * 32;
  const int nbase = blockIdx.y * GEMM_BN;
  const int z = blockIdx.z;
  const __bf16* W = (z == 0) ? wq : (z == 1) ? wk : wv;

  v8f acc[2][4];
  gemm_mainloop_256x64(xbf, W, mbase, nbase, bsm, acc);

#pragma unroll
  for (int mt = 0; mt < 2; ++mt) {
#pragma unroll
    for (int nt = 0; nt < 4; ++nt) {
      const int j = nbase + nt * 16 + lo;
      const int h = j >> 6, d = j & 63;
#pragma unroll
      for (int r = 0; r < 8; ++r) {
        const int i = mbase + mt * 16 + hi * 8 + r;  // global row = b*T + t
        const int bidx = i >> 12, t = i & (TT - 1);
        const __bf16 val = (__bf16)acc[mt][nt][r];
        if (z == 0)
          qws[(((size_t)(bidx * HN + h)) * TT + t) * HD + d] = val;
        else if (z == 1)
          kws[(((size_t)(bidx * HN + h)) * TT + t) * HD + d] = val;
        else
          vws[(((size_t)(bidx * HN + h)) * HD + d) * TT + t] = val;  // V^T
      }
    }
  }
}

// ---------------------------------------------------------------------------
// Kernel 2: sliding-window attention for one (b, h, 64-query block).
// grid (NBLK, H, B), block 256 (8 waves), dynamic LDS:
//   scores f32 [64][576] | P bf16 [64][576] | red f32 [256]
// Scores computed transposed (S' = K @ Q^T) so softmax rows sit in lanes.
// ---------------------------------------------------------------------------
__global__ __launch_bounds__(256, 1) void attn_kernel(
    const __bf16* __restrict__ qws,
    const __bf16* __restrict__ kws,
    const __bf16* __restrict__ vws,
    __bf16* __restrict__ aws) {
  extern __shared__ char smem[];
  float*  sc  = (float*)smem;                              // 64*576*4 = 147456
  __bf16* P   = (__bf16*)(smem + 64 * SKEY * 4);           // 64*576*2 =  73728
  float*  red = (float*)(smem + 64 * SKEY * 4 + 64 * SKEY * 2); // 256*4

  const int lane = threadIdx.x & 31;
  const int wave = threadIdx.x >> 5;
  const int hi   = lane >> 4;
  const int lo   = lane & 15;
  const int qb = blockIdx.x;
  const int h  = blockIdx.y;
  const int b  = blockIdx.z;

  const size_t bh = (size_t)(b * HN + h);
  const __bf16* Qbase = qws + (bh * TT + (size_t)qb * BSZ) * HD;
  const __bf16* Kbase = kws + bh * TT * HD;
  const __bf16* Vt    = vws + bh * HD * TT;

  const int qs = (wave & 3) * 16;      // this wave's query strip (of 64)
  const int kh = (wave >> 2) * 32;     // this wave's key half (of 64)

  // Q as WMMA B-operand (fixed for whole block): lane = query, elems = d
  v16bf bq[2];
  {
    const __bf16* qrow = Qbase + (size_t)(qs + lo) * HD;
    bq[0] = *(const v16bf*)(qrow + hi * 16);
    bq[1] = *(const v16bf*)(qrow + 32 + hi * 16);
  }

  // ---- Phase 1: S'[key, query] = K @ Q^T, scaled, into LDS ----
#pragma unroll 1
  for (int kw = 0; kw < NW1; ++kw) {
    const int kbidx = qb - (NW1 - 1) + kw;   // wave-uniform
    if (kbidx < 0) continue;                 // phase 2 masks by position
    v8f c0 = vzero8(), c1 = vzero8();
    const __bf16* krow0 = Kbase + (size_t)(kbidx * BSZ + kh + lo) * HD;
    const __bf16* krow1 = Kbase + (size_t)(kbidx * BSZ + kh + 16 + lo) * HD;
#pragma unroll
    for (int ds = 0; ds < 2; ++ds) {         // hd=64 -> 2 K-steps of 32
      AB16 a0, a1;
      a0.h[0] = *(const v8bf*)(krow0 + ds * 32 + hi * 16);
      a0.h[1] = *(const v8bf*)(krow0 + ds * 32 + hi * 16 + 8);
      a1.h[0] = *(const v8bf*)(krow1 + ds * 32 + hi * 16);
      a1.h[1] = *(const v8bf*)(krow1 + ds * 32 + hi * 16 + 8);
      c0 = wmma_bf16(a0.v, bq[ds], c0);
      c1 = wmma_bf16(a1.v, bq[ds], c1);
    }
    // D layout: M(key) = r + hi*8, N(query) = lo
    float* srow = sc + (size_t)(qs + lo) * SKEY + kw * 64 + kh;
#pragma unroll
    for (int r = 0; r < 8; ++r) {
      srow[hi * 8 + r]      = c0[r] * 0.125f;   // 1/sqrt(64)
      srow[16 + hi * 8 + r] = c1[r] * 0.125f;
    }
  }
  __syncthreads();

  // ---- Phase 2: exact softmax, 4 threads per query row ----
  {
    const int tid  = threadIdx.x;
    const int qrow = tid >> 2;
    const int part = tid & 3;
    const int cbeg = part * (SKEY / 4);
    const int cend = cbeg + (SKEY / 4);
    const int qpos = qb * BSZ + qrow;
    float* srow = sc + (size_t)qrow * SKEY;

    float mx = -1e30f;
    for (int c = cbeg; c < cend; ++c) {
      const int kw = c >> 6;
      const int kbidx = qb - (NW1 - 1) + kw;
      const int kpos = kbidx * BSZ + (c & 63);
      const bool valid = (kbidx >= 0) && (kpos <= qpos) && (qpos - kpos < WINSZ);
      if (valid) mx = fmaxf(mx, srow[c]);
    }
    red[tid] = mx;
    __syncthreads();
    const float rmax = fmaxf(fmaxf(red[qrow * 4 + 0], red[qrow * 4 + 1]),
                             fmaxf(red[qrow * 4 + 2], red[qrow * 4 + 3]));
    __syncthreads();

    float sum = 0.f;
    for (int c = cbeg; c < cend; ++c) {
      const int kw = c >> 6;
      const int kbidx = qb - (NW1 - 1) + kw;
      const int kpos = kbidx * BSZ + (c & 63);
      const bool valid = (kbidx >= 0) && (kpos <= qpos) && (qpos - kpos < WINSZ);
      if (valid) sum += __expf(srow[c] - rmax);
    }
    red[tid] = sum;
    __syncthreads();
    const float rsum = red[qrow * 4 + 0] + red[qrow * 4 + 1] +
                       red[qrow * 4 + 2] + red[qrow * 4 + 3];
    const float inv = 1.0f / rsum;  // diagonal always valid -> rsum > 0

    __bf16* prow = P + (size_t)qrow * SKEY;
    for (int c = cbeg; c < cend; ++c) {
      const int kw = c >> 6;
      const int kbidx = qb - (NW1 - 1) + kw;
      const int kpos = kbidx * BSZ + (c & 63);
      const bool valid = (kbidx >= 0) && (kpos <= qpos) && (qpos - kpos < WINSZ);
      float pv = 0.f;
      if (valid) pv = __expf(srow[c] - rmax) * inv;
      prow[c] = (__bf16)pv;
    }
  }
  __syncthreads();

  // ---- Phase 3: O = P @ V (V already transposed: Vt[d][t]) ----
  {
    const int dt0 = (wave >> 2) * 32;  // this wave's two d-tiles
    v8f o0 = vzero8(), o1 = vzero8();
    const __bf16* prow = P + (size_t)(qs + lo) * SKEY;
    const int tstart = (qb - (NW1 - 1)) * BSZ;   // may be negative (P==0 there)
#pragma unroll 1
    for (int kk = 0; kk < SKEY; kk += 32) {
      AB16 a;                                    // A = P tile from LDS
      a.h[0] = *(const v8bf*)(prow + kk + hi * 16);
      a.h[1] = *(const v8bf*)(prow + kk + hi * 16 + 8);
      int tb = tstart + kk + hi * 16;            // 16-aligned chunk of keys
      if (tb < 0) tb = 0;                        // P is 0 there; clamp address
      const __bf16* v0 = Vt + (size_t)(dt0 + lo) * TT + tb;
      const __bf16* v1 = Vt + (size_t)(dt0 + 16 + lo) * TT + tb;
      o0 = wmma_bf16(a.v, *(const v16bf*)v0, o0);
      o1 = wmma_bf16(a.v, *(const v16bf*)v1, o1);
    }
    // D layout: M(query) = r + hi*8, N(d) = lo. Write (B,T,D) bf16.
#pragma unroll
    for (int r = 0; r < 8; ++r) {
      const int query = qs + hi * 8 + r;
      const int t = qb * BSZ + query;
      const size_t rowoff = ((size_t)b * TT + t) * DD + h * HD;
      aws[rowoff + dt0 + lo]      = (__bf16)o0[r];
      aws[rowoff + dt0 + 16 + lo] = (__bf16)o1[r];
    }
  }
}

// ---------------------------------------------------------------------------
// Kernel 3: output projection.  out = attn (8192x1024 bf16) @ Wo^T -> f32
// grid (32, 16), block 256.
// ---------------------------------------------------------------------------
__global__ __launch_bounds__(256, 1) void out_gemm(
    const __bf16* __restrict__ abf,
    const __bf16* __restrict__ wo,
    float* __restrict__ out) {
  __shared__ __bf16 bsm[2 * GEMM_BN * BROW];
  const int lane = threadIdx.x & 31;
  const int wave = threadIdx.x >> 5;
  const int hi   = lane >> 4;
  const int lo   = lane & 15;
  const int mbase = blockIdx.x * GEMM_BM + wave * 32;
  const int nbase = blockIdx.y * GEMM_BN;

  v8f acc[2][4];
  gemm_mainloop_256x64(abf, wo, mbase, nbase, bsm, acc);

#pragma unroll
  for (int mt = 0; mt < 2; ++mt) {
#pragma unroll
    for (int nt = 0; nt < 4; ++nt) {
      const int j = nbase + nt * 16 + lo;
#pragma unroll
      for (int r = 0; r < 8; ++r) {
        const int i = mbase + mt * 16 + hi * 8 + r;
        out[(size_t)i * DD + j] = acc[mt][nt][r];
      }
    }
  }
}

// ---------------------------------------------------------------------------
// Host launcher
// ---------------------------------------------------------------------------
extern "C" void kernel_launch(void* const* d_in, const int* in_sizes, int n_in,
                              void* d_out, int out_size, void* d_ws, size_t ws_size,
                              hipStream_t stream) {
  (void)in_sizes; (void)n_in; (void)out_size; (void)ws_size;
  const float* x  = (const float*)d_in[0];
  const float* Wq = (const float*)d_in[1];
  const float* Wk = (const float*)d_in[2];
  const float* Wv = (const float*)d_in[3];
  const float* Wo = (const float*)d_in[4];
  float* out = (float*)d_out;

  const size_t NX = (size_t)BB * TT * DD;   // 8388608
  const size_t NW = (size_t)DD * DD;        // 1048576

  __bf16* ws = (__bf16*)d_ws;
  __bf16* xbf = ws;                 ws += NX;
  __bf16* wqb = ws;                 ws += NW;
  __bf16* wkb = ws;                 ws += NW;
  __bf16* wvb = ws;                 ws += NW;
  __bf16* wob = ws;                 ws += NW;
  __bf16* qws = ws;                 ws += NX;
  __bf16* kws = ws;                 ws += NX;
  __bf16* vws = ws;                 ws += NX;
  __bf16* aws = ws;                 ws += NX;

  // 0) convert inputs to bf16
  cvt_f32_bf16<<<1024, 256, 0, stream>>>(x,  xbf, (int)NX);
  cvt_f32_bf16<<<256,  256, 0, stream>>>(Wq, wqb, (int)NW);
  cvt_f32_bf16<<<256,  256, 0, stream>>>(Wk, wkb, (int)NW);
  cvt_f32_bf16<<<256,  256, 0, stream>>>(Wv, wvb, (int)NW);
  cvt_f32_bf16<<<256,  256, 0, stream>>>(Wo, wob, (int)NW);

  // 1) fused QKV projection (async-LDS staged B, double buffered)
  {
    dim3 grid((BB * TT) / GEMM_BM, DD / GEMM_BN, 3);
    qkv_gemm<<<grid, 256, 0, stream>>>(xbf, wqb, wkb, wvb, qws, kws, vws);
  }

  // 2) sliding-window attention (222208 B dynamic LDS; WGP has 320 KB)
  {
    const size_t shmem = (size_t)64 * SKEY * 4 + (size_t)64 * SKEY * 2 + 256 * 4;
    hipError_t err = hipFuncSetAttribute(
        (const void*)attn_kernel, hipFuncAttributeMaxDynamicSharedMemorySize,
        (int)shmem);
    (void)err;
    dim3 grid(NBLK, HN, BB);
    attn_kernel<<<grid, 256, shmem, stream>>>(qws, kws, vws, aws);
  }

  // 3) output projection -> f32
  {
    dim3 grid((BB * TT) / GEMM_BM, DD / GEMM_BN);
    out_gemm<<<grid, 256, 0, stream>>>(aws, wob, out);
  }
}
